// CoordinatedActionExecutor_704374637170
// MI455X (gfx1250) — matrix operand order repaired and checked by reading, not compiled, not run on hardware
//
#include <hip/hip_runtime.h>
#include <hip/hip_bf16.h>

// ---------------------------------------------------------------------------
// CDNA5 (gfx1250) implementation: bf16 WMMA GEMMs + persistent-LDS LSTM scan.
// ---------------------------------------------------------------------------

typedef __attribute__((ext_vector_type(16))) __bf16 v16bf;
typedef __attribute__((ext_vector_type(8)))  float  v8f;

#define HDIM 256
#define LSTR 264   // padded LDS row stride (halves): 528B row -> conflict-friendly

static __device__ __forceinline__ unsigned short f32_to_bf16(float f) {
  unsigned int x = __float_as_uint(f);
  x += 0x7FFFu + ((x >> 16) & 1u);          // round-to-nearest-even
  return (unsigned short)(x >> 16);
}
static __device__ __forceinline__ float bf_lo(unsigned int w) { return __uint_as_float(w << 16); }
static __device__ __forceinline__ float bf_hi(unsigned int w) { return __uint_as_float(w & 0xFFFF0000u); }

union FragU { v16bf v; uint4 u[2]; };

// A fragment (16x32 bf16, 16-bit A layout): p points at (row, Kbase) where
// Kbase = k0*32 + (lane>=16 ? 8 : 0); halves 0-7 = K+0..7, halves 8-15 = K+16..23.
static __device__ __forceinline__ v16bf load_a_frag(const unsigned short* p) {
  FragU f;
  f.u[0] = *(const uint4*)(p);
  f.u[1] = *(const uint4*)(p + 16);
  return f.v;
}
// B fragment from pre-swizzled weights: [(ntile*8 + k0)*32 + lane][16 halves] contiguous.
static __device__ __forceinline__ v16bf load_b_frag(const unsigned short* Wsw, int ntile, int k0, int lane) {
  const unsigned short* p = Wsw + (size_t)(((ntile * 8 + k0) * 32) + lane) * 16;
  FragU f;
  f.u[0] = *(const uint4*)(p);
  f.u[1] = *(const uint4*)(p + 8);
  return f.v;
}
static __device__ __forceinline__ v8f wmma_bf16(v16bf a, v16bf b, v8f c) {
  return __builtin_amdgcn_wmma_f32_16x16x32_bf16(false, a, false, b, (short)0, c, false, false);
}
static __device__ __forceinline__ float sigmoidf_(float x) { return 1.0f / (1.0f + __expf(-x)); }

// ---------------------------------------------------------------------------
// Prep kernels
// ---------------------------------------------------------------------------

// inv[agent] = group  (honors agent_groups instead of assuming arange)
__global__ void k_inv(const int* __restrict__ groups, int* __restrict__ inv, int n) {
  int e = blockIdx.x * blockDim.x + threadIdx.x;
  if (e < n) inv[groups[e]] = e >> 5;
}

// generic f32 -> bf16 (row-major preserved); used for W_hh
__global__ void k_cvt(const float* __restrict__ src, unsigned short* __restrict__ dst, int n) {
  int i = blockIdx.x * blockDim.x + threadIdx.x;
  if (i < n) dst[i] = f32_to_bf16(src[i]);
}

// swizzle row-major f32 W[K][N] into per-lane WMMA B-fragment order (bf16).
// dst[((ntile*KT + k0)*32 + lane)*16 + h] = W[k0*32 + (lane>=16?8:0) + (h&7) + (h>=8?16:0)][ntile*16 + (lane&15)]
__global__ void k_swz(const float* __restrict__ W, unsigned short* __restrict__ Wsw, int K, int N) {
  int idx = blockIdx.x * blockDim.x + threadIdx.x;
  if (idx >= K * N) return;
  int KT = K >> 5;
  int h    = idx & 15;
  int lane = (idx >> 4) & 31;
  int t    = idx >> 9;
  int k0    = t % KT;
  int ntile = t / KT;
  int col = ntile * 16 + (lane & 15);
  int kk  = k0 * 32 + ((lane >= 16) ? 8 : 0) + (h & 7) + ((h >= 8) ? 16 : 0);
  Wsw[idx] = f32_to_bf16(W[kk * N + col]);
}

// goal_emb = relu(goal @ W_goal + b_goal) -> arows row 1024 (bf16); zero pad rows 1025..1039
__global__ void k_goal(const float* __restrict__ goal, const float* __restrict__ Wg,
                       const float* __restrict__ bg, unsigned short* __restrict__ arows) {
  int h = threadIdx.x;                       // 256 threads
  float acc = bg[h];
  for (int k = 0; k < HDIM; ++k) acc = fmaf(goal[k], Wg[k * HDIM + h], acc);
  acc = fmaxf(acc, 0.0f);
  arows[1024 * HDIM + h] = f32_to_bf16(acc);
  for (int r = 1025; r < 1040; ++r) arows[r * HDIM + h] = 0;
}

// ---------------------------------------------------------------------------
// Fused per-group GCN x2 + mean-pool. One workgroup (8 waves) per group.
//   out = relu((colsum(xW) + xW)/33 + b)  twice;  arows[g] = colmean(h2)
// ---------------------------------------------------------------------------
__global__ __launch_bounds__(256) void k_gcn(const float* __restrict__ x,
                                             const int* __restrict__ groups,
                                             const unsigned short* __restrict__ W1sw,
                                             const float* __restrict__ b1,
                                             const unsigned short* __restrict__ W2sw,
                                             const float* __restrict__ b2,
                                             unsigned short* __restrict__ arows) {
  __shared__ __align__(16) unsigned short Xa[32 * LSTR];
  __shared__ __align__(16) unsigned short Hb[32 * LSTR];
  const int g = blockIdx.x, tid = threadIdx.x;
  const int wave = tid >> 5, lane = tid & 31, lr = lane & 15, half = lane >> 4;

  // gather group rows (via agent_groups) -> bf16 LDS, A-operand layout (row-major, padded)
  for (int i = tid; i < 32 * HDIM; i += 256) {
    int r = i >> 8, c = i & 255;
    int a = groups[g * 32 + r];
    Xa[r * LSTR + c] = f32_to_bf16(x[(size_t)a * HDIM + c]);
  }
  __syncthreads();

  // ---- layer 1 ----
  {
    v8f acc[2][2] = {};
    for (int k0 = 0; k0 < 8; ++k0) {
      const int koff = k0 * 32 + half * 8;
      v16bf a0 = load_a_frag(&Xa[(0  + lr) * LSTR + koff]);
      v16bf a1 = load_a_frag(&Xa[(16 + lr) * LSTR + koff]);
      v16bf bA = load_b_frag(W1sw, 2 * wave,     k0, lane);
      v16bf bB = load_b_frag(W1sw, 2 * wave + 1, k0, lane);
      acc[0][0] = wmma_bf16(a0, bA, acc[0][0]);
      acc[1][0] = wmma_bf16(a1, bA, acc[1][0]);
      acc[0][1] = wmma_bf16(a0, bB, acc[0][1]);
      acc[1][1] = wmma_bf16(a1, bB, acc[1][1]);
    }
    #pragma unroll
    for (int ns = 0; ns < 2; ++ns) {
      int col = wave * 32 + ns * 16 + lr;
      float bias = b1[col];
      float cs = 0.0f;
      #pragma unroll
      for (int r = 0; r < 8; ++r) cs += acc[0][ns][r] + acc[1][ns][r];
      cs += __shfl_xor(cs, 16, 32);            // full 32-row column sum
      #pragma unroll
      for (int m = 0; m < 2; ++m)
        #pragma unroll
        for (int r = 0; r < 8; ++r) {
          float v = (cs + acc[m][ns][r]) * (1.0f / 33.0f) + bias;
          v = fmaxf(v, 0.0f);
          Hb[(m * 16 + half * 8 + r) * LSTR + col] = f32_to_bf16(v);
        }
    }
  }
  __syncthreads();

  // ---- layer 2 + mean pool (per-row h2 never needs to be materialized) ----
  {
    v8f acc[2][2] = {};
    for (int k0 = 0; k0 < 8; ++k0) {
      const int koff = k0 * 32 + half * 8;
      v16bf a0 = load_a_frag(&Hb[(0  + lr) * LSTR + koff]);
      v16bf a1 = load_a_frag(&Hb[(16 + lr) * LSTR + koff]);
      v16bf bA = load_b_frag(W2sw, 2 * wave,     k0, lane);
      v16bf bB = load_b_frag(W2sw, 2 * wave + 1, k0, lane);
      acc[0][0] = wmma_bf16(a0, bA, acc[0][0]);
      acc[1][0] = wmma_bf16(a1, bA, acc[1][0]);
      acc[0][1] = wmma_bf16(a0, bB, acc[0][1]);
      acc[1][1] = wmma_bf16(a1, bB, acc[1][1]);
    }
    #pragma unroll
    for (int ns = 0; ns < 2; ++ns) {
      int col = wave * 32 + ns * 16 + lr;
      float bias = b2[col];
      float cs = 0.0f;
      #pragma unroll
      for (int r = 0; r < 8; ++r) cs += acc[0][ns][r] + acc[1][ns][r];
      cs += __shfl_xor(cs, 16, 32);
      float ps = 0.0f;
      #pragma unroll
      for (int m = 0; m < 2; ++m)
        #pragma unroll
        for (int r = 0; r < 8; ++r) {
          float v = (cs + acc[m][ns][r]) * (1.0f / 33.0f) + bias;
          ps += fmaxf(v, 0.0f);
        }
      ps += __shfl_xor(ps, 16, 32);            // sum over all 32 rows
      if (half == 0) arows[g * HDIM + col] = f32_to_bf16(ps * (1.0f / 32.0f));
    }
  }
}

// ---------------------------------------------------------------------------
// zs[g] = arows[g] @ W_ih + b_lstm   for the 1025 DISTINCT seq rows only.
// grid (65 mtiles, 8), 8 waves/block -> one 16x16 tile per wave.
// ---------------------------------------------------------------------------
__global__ __launch_bounds__(256) void k_zs(const unsigned short* __restrict__ arows,
                                            const unsigned short* __restrict__ Wihsw,
                                            const float* __restrict__ b_lstm,
                                            float* __restrict__ zs) {
  const int wave = threadIdx.x >> 5, lane = threadIdx.x & 31, lr = lane & 15, half = lane >> 4;
  const int mtile = blockIdx.x;
  const int ntile = blockIdx.y * 8 + wave;
  v8f acc = {};
  for (int k0 = 0; k0 < 8; ++k0) {
    const unsigned short* p = arows + (size_t)(mtile * 16 + lr) * HDIM + k0 * 32 + half * 8;
    v16bf a = load_a_frag(p);
    v16bf b = load_b_frag(Wihsw, ntile, k0, lane);
    acc = wmma_bf16(a, b, acc);
  }
  int col = ntile * 16 + lr;
  float bias = b_lstm[col];
  #pragma unroll
  for (int r = 0; r < 8; ++r) {
    int row = mtile * 16 + half * 8 + r;
    if (row < 1025) zs[(size_t)row * 1024 + col] = acc[r] + bias;
  }
}

// ---------------------------------------------------------------------------
// Persistent LSTM scan: one workgroup (256 thr / 8 waves), 32769 steps.
// Thread j owns gate columns 4j..4j+3 for the matvec (coalesced bf16 uint2 per k)
// and cell j for the state update (c in a register). W_hh stays hot in L2.
// ---------------------------------------------------------------------------
__global__ __launch_bounds__(256) void k_lstm(const float* __restrict__ zs,
                                              const unsigned short* __restrict__ whh,
                                              const int* __restrict__ inv,
                                              unsigned short* __restrict__ hs) {
  __shared__ __align__(16) float hL[HDIM];
  __shared__ __align__(16) float zL[1024];
  const int j = threadIdx.x;                   // 0..255
  float c = 0.0f;
  hL[j] = 0.0f;
  __syncthreads();

  const unsigned short* wp = whh + 4 * j;      // column base within each row
  for (int t = 0; t <= 32768; ++t) {
    const int zrow = (t < 32768) ? inv[t] : 1024;
    const float* zb = zs + (size_t)zrow * 1024 + 4 * j;
    float a0 = zb[0], a1 = zb[1], a2 = zb[2], a3 = zb[3];
    #pragma unroll 4
    for (int k = 0; k < HDIM; ++k) {
      float hk = hL[k];                        // LDS broadcast
      uint2 w = *(const uint2*)(wp + (size_t)k * 1024);
      a0 = fmaf(hk, bf_lo(w.x), a0);
      a1 = fmaf(hk, bf_hi(w.x), a1);
      a2 = fmaf(hk, bf_lo(w.y), a2);
      a3 = fmaf(hk, bf_hi(w.y), a3);
    }
    *(float4*)&zL[4 * j] = make_float4(a0, a1, a2, a3);
    __syncthreads();
    // gates (order i, f, g, o)
    float zi = zL[j], zf = zL[j + 256], zg = zL[j + 512], zo = zL[j + 768];
    c = sigmoidf_(zf) * c + sigmoidf_(zi) * tanhf(zg);
    float h = sigmoidf_(zo) * tanhf(c);
    hL[j] = h;
    hs[(size_t)t * HDIM + j] = f32_to_bf16(h);
    __syncthreads();
  }
}

// ---------------------------------------------------------------------------
// out = softmax(hs @ W_act + b_act). 2 mtiles x 4 ntiles per block (8 waves),
// LDS stage for the cross-wave row softmax over 64 logits.
// ---------------------------------------------------------------------------
__global__ __launch_bounds__(256) void k_proj(const unsigned short* __restrict__ hs,
                                              const unsigned short* __restrict__ Wactsw,
                                              const float* __restrict__ b_act,
                                              float* __restrict__ out) {
  __shared__ float stage[32][65];
  const int wave = threadIdx.x >> 5, lane = threadIdx.x & 31, lr = lane & 15, half = lane >> 4;
  const int mt = blockIdx.x * 2 + (wave >> 2);
  const int nt = wave & 3;
  v8f acc = {};
  for (int k0 = 0; k0 < 8; ++k0) {
    const unsigned short* p = hs + (size_t)(mt * 16 + lr) * HDIM + k0 * 32 + half * 8;
    v16bf a = load_a_frag(p);
    v16bf b = load_b_frag(Wactsw, nt, k0, lane);
    acc = wmma_bf16(a, b, acc);
  }
  int col = nt * 16 + lr;
  float bias = b_act[col];
  int rbase = (wave >> 2) * 16 + half * 8;
  #pragma unroll
  for (int r = 0; r < 8; ++r) stage[rbase + r][col] = acc[r] + bias;
  __syncthreads();

  if (threadIdx.x < 32) {
    int row = blockIdx.x * 32 + threadIdx.x;
    if (row < 32769) {
      float* s = stage[threadIdx.x];
      float m = s[0];
      for (int i = 1; i < 64; ++i) m = fmaxf(m, s[i]);
      float sum = 0.0f;
      for (int i = 0; i < 64; ++i) { float e = __expf(s[i] - m); s[i] = e; sum += e; }
      float is = 1.0f / sum;
      float* o = out + (size_t)row * 64;
      for (int i = 0; i < 64; ++i) o[i] = s[i] * is;
    }
  }
}

// ---------------------------------------------------------------------------
extern "C" void kernel_launch(void* const* d_in, const int* in_sizes, int n_in,
                              void* d_out, int out_size, void* d_ws, size_t ws_size,
                              hipStream_t stream) {
  (void)in_sizes; (void)n_in; (void)out_size; (void)ws_size;
  const float* agent_state = (const float*)d_in[0];
  const float* goal_state  = (const float*)d_in[1];
  const int*   groups      = (const int*)d_in[2];
  const float* W_goal = (const float*)d_in[3];
  const float* b_goal = (const float*)d_in[4];
  const float* W_g1   = (const float*)d_in[5];
  const float* b_g1   = (const float*)d_in[6];
  const float* W_g2   = (const float*)d_in[7];
  const float* b_g2   = (const float*)d_in[8];
  const float* W_ih   = (const float*)d_in[9];
  const float* W_hh   = (const float*)d_in[10];
  const float* b_lstm = (const float*)d_in[11];
  const float* W_act  = (const float*)d_in[12];
  const float* b_act  = (const float*)d_in[13];
  float* out = (float*)d_out;

  char* ws = (char*)d_ws;
  size_t off = 0;
  auto alloc = [&](size_t bytes) -> void* {
    void* p = ws + off;
    off = (off + bytes + 255) & ~(size_t)255;
    return p;
  };
  int*            inv    = (int*)alloc(32768 * sizeof(int));
  unsigned short* whh_bf = (unsigned short*)alloc((size_t)256 * 1024 * 2);
  unsigned short* w1sw   = (unsigned short*)alloc((size_t)256 * 256 * 2);
  unsigned short* w2sw   = (unsigned short*)alloc((size_t)256 * 256 * 2);
  unsigned short* wihsw  = (unsigned short*)alloc((size_t)256 * 1024 * 2);
  unsigned short* wactsw = (unsigned short*)alloc((size_t)256 * 64 * 2);
  unsigned short* arows  = (unsigned short*)alloc((size_t)1040 * 256 * 2);   // 1024 pooled + goal + pad
  float*          zs     = (float*)alloc((size_t)1025 * 1024 * sizeof(float));
  unsigned short* hsb    = (unsigned short*)alloc((size_t)32800 * 256 * 2);  // padded to 2050 mtiles

  k_inv <<<128, 256, 0, stream>>>(groups, inv, 32768);
  k_cvt <<<1024, 256, 0, stream>>>(W_hh, whh_bf, 256 * 1024);
  k_swz <<<256,  256, 0, stream>>>(W_g1,  w1sw,  256, 256);
  k_swz <<<256,  256, 0, stream>>>(W_g2,  w2sw,  256, 256);
  k_swz <<<1024, 256, 0, stream>>>(W_ih,  wihsw, 256, 1024);
  k_swz <<<64,   256, 0, stream>>>(W_act, wactsw, 256, 64);
  k_goal<<<1,    256, 0, stream>>>(goal_state, W_goal, b_goal, arows);
  k_gcn <<<1024, 256, 0, stream>>>(agent_state, groups, w1sw, b_g1, w2sw, b_g2, arows);
  k_zs  <<<dim3(65, 8), 256, 0, stream>>>(arows, wihsw, b_lstm, zs);
  k_lstm<<<1,    256, 0, stream>>>(zs, whh_bf, inv, hsb);
  k_proj<<<1025, 256, 0, stream>>>(hsb, wactsw, b_act, out);
}
